// DoubleStreamBlock_62388694942148
// MI455X (gfx1250) — matrix operand order repaired and checked by reading, not compiled, not run on hardware
//
#include <hip/hip_runtime.h>
#include <hip/hip_bf16.h>

// ---------------------------------------------------------------------------
// DoubleStreamBlock for MI455X (gfx1250): bf16 WMMA + async-to-LDS staging.
// ---------------------------------------------------------------------------

#define NH   16
#define HD   64
#define HID  1024
#define MLPD 4096
#define LIMG 2048
#define LTXT 512
#define LTOT 2560
#define EPSF 1e-6f

typedef __attribute__((ext_vector_type(16))) __bf16 v16bf;
typedef __attribute__((ext_vector_type(8)))  float  v8f;

union FragAB { v16bf v; unsigned u[8]; };
union FragC  { v8f   v; float    f[8]; };

__device__ __forceinline__ unsigned short f2bf(float x) {
    unsigned u = __float_as_uint(x);
    u += 0x7FFFu + ((u >> 16) & 1u);          // round-to-nearest-even
    return (unsigned short)(u >> 16);
}
__device__ __forceinline__ unsigned packbf2(float a, float b) {
    return (unsigned)f2bf(a) | ((unsigned)f2bf(b) << 16);
}

// gfx1250 async global->LDS copy (GLOBAL_LOAD_ASYNC_TO_LDS_B128, ASYNCcnt).
// ldsp: generic pointer into __shared__ (low 32 bits are the LDS offset).
__device__ __forceinline__ void async_b128(void* ldsp, const void* gptr) {
    unsigned loff = (unsigned)(size_t)ldsp;
    unsigned long long ga = (unsigned long long)(size_t)gptr;
    asm volatile("global_load_async_to_lds_b128 %0, %1, off"
                 :: "v"(loff), "v"(ga) : "memory");
}
__device__ __forceinline__ void wait_async0() {
    asm volatile("s_wait_asynccnt 0x0" ::: "memory");
}

// ---------------------------------------------------------------------------
// 0) weight convert+transpose: in[K][N] f32 -> out[N][K] bf16 (32x32 LDS tile)
// ---------------------------------------------------------------------------
__global__ __launch_bounds__(256) void convert_transpose(
    const float* __restrict__ in, unsigned short* __restrict__ out, int K, int N) {
    __shared__ float tile[32][33];
    const int k0 = blockIdx.x * 32, n0 = blockIdx.y * 32;
    const int tx = threadIdx.x & 31, ty = threadIdx.x >> 5;   // 32 x 8
#pragma unroll
    for (int i = 0; i < 4; ++i)
        tile[ty + 8 * i][tx] = in[(size_t)(k0 + ty + 8 * i) * N + n0 + tx];
    __syncthreads();
#pragma unroll
    for (int i = 0; i < 4; ++i)
        out[(size_t)(n0 + ty + 8 * i) * K + k0 + tx] = f2bf(tile[tx][ty + 8 * i]);
}

// ---------------------------------------------------------------------------
// 1) modulation GEMV: out[c] = dot(silu(vec), W[:,c]) + b[c]
// ---------------------------------------------------------------------------
__global__ __launch_bounds__(256) void mod_gemv(
    const float* __restrict__ vec, const float* __restrict__ w,
    const float* __restrict__ b, float* __restrict__ out, int ncols) {
    __shared__ float sv[HID];
    for (int i = threadIdx.x; i < HID; i += 256) {
        float x = vec[i];
        sv[i] = x / (1.f + __expf(-x));
    }
    __syncthreads();
    int c = blockIdx.x * 256 + threadIdx.x;
    if (c < ncols) {
        float acc = 0.f;
        for (int k = 0; k < HID; ++k) acc = fmaf(sv[k], w[(size_t)k * ncols + c], acc);
        out[c] = acc + b[c];
    }
}

// ---------------------------------------------------------------------------
// 2) LayerNorm + modulate -> bf16 (mod layout [6][HID]: sh1,sc1,g1,sh2,sc2,g2)
// ---------------------------------------------------------------------------
__global__ __launch_bounds__(256) void ln_mod(
    const float* __restrict__ x, const float* __restrict__ mod,
    int shIdx, int scIdx, unsigned short* __restrict__ out) {
    __shared__ float red[256];
    const float* row = x + (size_t)blockIdx.x * HID;
    float vals[4];
    float s = 0.f;
#pragma unroll
    for (int i = 0; i < 4; ++i) { vals[i] = row[threadIdx.x + i * 256]; s += vals[i]; }
    red[threadIdx.x] = s; __syncthreads();
    for (int o = 128; o > 0; o >>= 1) {
        if (threadIdx.x < o) red[threadIdx.x] += red[threadIdx.x + o];
        __syncthreads();
    }
    float mu = red[0] * (1.f / HID);
    __syncthreads();
    float v = 0.f;
#pragma unroll
    for (int i = 0; i < 4; ++i) { float d = vals[i] - mu; v += d * d; }
    red[threadIdx.x] = v; __syncthreads();
    for (int o = 128; o > 0; o >>= 1) {
        if (threadIdx.x < o) red[threadIdx.x] += red[threadIdx.x + o];
        __syncthreads();
    }
    float rstd = rsqrtf(red[0] * (1.f / HID) + EPSF);
    unsigned short* orow = out + (size_t)blockIdx.x * HID;
#pragma unroll
    for (int i = 0; i < 4; ++i) {
        int c = threadIdx.x + i * 256;
        float xm = (1.f + mod[scIdx * HID + c]) * ((vals[i] - mu) * rstd) + mod[shIdx * HID + c];
        orow[c] = f2bf(xm);
    }
}

// ---------------------------------------------------------------------------
// 3) bf16 WMMA GEMM, double-buffered async staging.
//    C[M,N] = A[M,K](bf16) @ Wt[N,K](bf16, pre-transposed)
//    block tile 128x64, BK=32; 8 waves (4x2), each wave 32x32.
//    mode 0: f32   mode 1: +bias,GELU->bf16   mode 2: res+gate*(C+bias)->f32
// ---------------------------------------------------------------------------
#define BM 128
#define BN 64
#define BK 32
#define APAD 20   // 32 bf16 = 16 uints, padded to 20 (80B, 16B-aligned rows)

__global__ __launch_bounds__(256) void gemm_bf16(
    const unsigned short* __restrict__ A, const unsigned short* __restrict__ Wt,
    const float* __restrict__ bias, const float* __restrict__ res,
    const float* __restrict__ gate, void* __restrict__ out,
    int M, int N, int K, int mode) {
    __shared__ __align__(16) unsigned As[2][BM][APAD];
    __shared__ __align__(16) unsigned Bs[2][BN][APAD];
    const int tid  = threadIdx.x;
    const int lane = tid & 31;
    const int wave = tid >> 5;
    const int m0 = blockIdx.x * BM;
    const int n0 = blockIdx.y * BN;
    const int wm = (wave >> 1) * 32;
    const int wn = (wave & 1) * 32;
    const int am = lane & 15;
    const int hh = lane >> 4;

    FragC acc[2][2];
#pragma unroll
    for (int i = 0; i < 2; ++i)
#pragma unroll
        for (int j = 0; j < 2; ++j)
#pragma unroll
            for (int r = 0; r < 8; ++r) acc[i][j].f[r] = 0.f;

    auto stage = [&](int buf, int k0) {
        // A tile: 128 rows x 64B = 512 x b128; 2 per thread
#pragma unroll
        for (int t = 0; t < 2; ++t) {
            int i = tid + t * 256;
            int r = i >> 2, p = i & 3;
            async_b128(&As[buf][r][p * 4], A + (size_t)(m0 + r) * K + k0 + p * 8);
        }
        // B tile: 64 rows x 64B = 256 x b128; 1 per thread
        {
            int r = tid >> 2, p = tid & 3;
            async_b128(&Bs[buf][r][p * 4], Wt + (size_t)(n0 + r) * K + k0 + p * 8);
        }
    };

    const int nt = K / BK;
    stage(0, 0);
    wait_async0();
    __syncthreads();

    for (int kt = 0; kt < nt; ++kt) {
        const int cur = kt & 1;
        if (kt + 1 < nt) stage(cur ^ 1, (kt + 1) * BK);

        FragAB afr[2], bfr[2];
#pragma unroll
        for (int t = 0; t < 2; ++t) {
#pragma unroll
            for (int v = 0; v < 8; ++v) {
                int kpa = ((v < 4) ? 0 : 8) + hh * 4 + (v & 3);
                afr[t].u[v] = As[cur][wm + t * 16 + am][kpa];
            }
#pragma unroll
            for (int v = 0; v < 8; ++v) {
                int kpb = ((v < 4) ? 0 : 8) + hh * 4 + (v & 3);
                bfr[t].u[v] = Bs[cur][wn + t * 16 + am][kpb];
            }
        }
#pragma unroll
        for (int i = 0; i < 2; ++i)
#pragma unroll
            for (int j = 0; j < 2; ++j)
                acc[i][j].v = __builtin_amdgcn_wmma_f32_16x16x32_bf16(
                    false, afr[i].v, false, bfr[j].v, (short)0, acc[i][j].v, false, false);

        wait_async0();
        __syncthreads();
    }

    // epilogue
#pragma unroll
    for (int i = 0; i < 2; ++i)
#pragma unroll
        for (int j = 0; j < 2; ++j)
#pragma unroll
            for (int r = 0; r < 8; ++r) {
                int row = m0 + wm + i * 16 + r + 8 * hh;
                int col = n0 + wn + j * 16 + am;
                float c = acc[i][j].f[r];
                size_t idx = (size_t)row * N + col;
                if (mode == 0) {
                    ((float*)out)[idx] = c;
                } else if (mode == 1) {
                    c += bias[col];
                    float g = 0.5f * c * (1.f + tanhf(0.7978845608f * (c + 0.044715f * c * c * c)));
                    ((unsigned short*)out)[idx] = f2bf(g);
                } else {
                    ((float*)out)[idx] = res[idx] + gate[col] * (c + bias[col]);
                }
            }
}

// ---------------------------------------------------------------------------
// 4) qkv post: RMS-norm q,k per (token,head); pack Q/K/V bf16 head-major.
// ---------------------------------------------------------------------------
__global__ __launch_bounds__(256) void qkv_post(
    const float* __restrict__ qkv, const float* __restrict__ qs,
    const float* __restrict__ ks, unsigned short* __restrict__ Qg,
    unsigned short* __restrict__ Kg, unsigned short* __restrict__ Vg,
    int L, int tokOff) {
    int idx = blockIdx.x * 8 + (threadIdx.x >> 5);
    int lane = threadIdx.x & 31;
    if (idx >= L * NH) return;
    int t = idx / NH, h = idx % NH;
    const float* base = qkv + (size_t)t * (3 * HID);
    float q0 = base[h * HD + lane],           q1 = base[h * HD + lane + 32];
    float k0 = base[HID + h * HD + lane],     k1 = base[HID + h * HD + lane + 32];
    float v0 = base[2 * HID + h * HD + lane], v1 = base[2 * HID + h * HD + lane + 32];
    float sq = q0 * q0 + q1 * q1;
    float sk = k0 * k0 + k1 * k1;
#pragma unroll
    for (int o = 16; o > 0; o >>= 1) {
        sq += __shfl_xor(sq, o, 32);
        sk += __shfl_xor(sk, o, 32);
    }
    float rq = rsqrtf(sq * (1.f / HD) + EPSF);
    float rk = rsqrtf(sk * (1.f / HD) + EPSF);
    size_t ob = ((size_t)h * LTOT + (tokOff + t)) * HD;
    Qg[ob + lane]      = f2bf(q0 * rq * qs[lane]);
    Qg[ob + lane + 32] = f2bf(q1 * rq * qs[lane + 32]);
    Kg[ob + lane]      = f2bf(k0 * rk * ks[lane]);
    Kg[ob + lane + 32] = f2bf(k1 * rk * ks[lane + 32]);
    Vg[ob + lane]      = f2bf(v0);
    Vg[ob + lane + 32] = f2bf(v1);
}

// ---------------------------------------------------------------------------
// 5) Flash attention: per (head, 64-query tile); KV tiles of 64.
// ---------------------------------------------------------------------------
#define QT 64
#define KT 64
#define QPAD 36  // 32 uints + 4 pad (144B rows, 16B-aligned)

__global__ __launch_bounds__(256) void attn_kernel(
    const unsigned short* __restrict__ Qg, const unsigned short* __restrict__ Kg,
    const unsigned short* __restrict__ Vg, unsigned short* __restrict__ Og) {
    __shared__ __align__(16) unsigned Qs[QT][QPAD];   // [q][d-pair]
    __shared__ __align__(16) unsigned Ks[KT][QPAD];   // [kv][d-pair]
    __shared__ __align__(16) unsigned Vt[HD][QPAD];   // [d][kv-pair] (transposed)
    __shared__ float    Sf[QT][KT + 2];
    __shared__ unsigned Ps[QT][QPAD];                 // [q][kv-pair]
    __shared__ float    mrow[QT], lrow[QT], srow[QT];

    const int h    = blockIdx.y;
    const int q0   = blockIdx.x * QT;
    const int tid  = threadIdx.x;
    const int lane = tid & 31;
    const int wave = tid >> 5;
    const int am = lane & 15;
    const int hh = lane >> 4;
    const int wr = (wave >> 1) * 16;     // this wave's O rows
    const int wc = (wave & 1) * 32;      // this wave's O cols

    // async-stage Q tile (64 rows x 128B = 512 x b128; 2 per thread)
    const unsigned short* qsrc = Qg + ((size_t)h * LTOT + q0) * HD;
#pragma unroll
    for (int t = 0; t < 2; ++t) {
        int i = tid + t * 256;
        int r = i >> 3, p = i & 7;
        async_b128(&Qs[r][p * 4], qsrc + (size_t)r * HD + p * 8);
    }
    if (tid < QT) { mrow[tid] = -1e30f; lrow[tid] = 0.f; }

    FragC o[2];
#pragma unroll
    for (int j = 0; j < 2; ++j)
#pragma unroll
        for (int r = 0; r < 8; ++r) o[j].f[r] = 0.f;
    wait_async0();
    __syncthreads();

    for (int kt = 0; kt < LTOT / KT; ++kt) {
        const int kbase = kt * KT;
        // async-stage K tile
        const unsigned short* ksrc = Kg + ((size_t)h * LTOT + kbase) * HD;
#pragma unroll
        for (int t = 0; t < 2; ++t) {
            int i = tid + t * 256;
            int r = i >> 3, p = i & 7;
            async_b128(&Ks[r][p * 4], ksrc + (size_t)r * HD + p * 8);
        }
        // stage V transposed: [d][kv-pair], wide b64 reads
        const unsigned short* vsrc = Vg + ((size_t)h * LTOT + kbase) * HD;
#pragma unroll
        for (int t = 0; t < 2; ++t) {
            int i = tid + t * 256;            // 512 = 32 pairs x 16 d-groups
            int tp = i >> 4, dg = (i & 15) * 4;
            const unsigned* p0 = (const unsigned*)(vsrc + (size_t)(2 * tp) * HD + dg);
            const unsigned* p1 = (const unsigned*)(vsrc + (size_t)(2 * tp + 1) * HD + dg);
            unsigned a0 = p0[0], a1 = p0[1], b0 = p1[0], b1 = p1[1];
            Vt[dg + 0][tp] = (a0 & 0xFFFFu) | (b0 << 16);
            Vt[dg + 1][tp] = (a0 >> 16) | (b0 & 0xFFFF0000u);
            Vt[dg + 2][tp] = (a1 & 0xFFFFu) | (b1 << 16);
            Vt[dg + 3][tp] = (a1 >> 16) | (b1 & 0xFFFF0000u);
        }
        wait_async0();
        __syncthreads();

        // S = Q @ K^T : 16 tiles of 16x16; each wave does 2
        for (int tt = wave; tt < 16; tt += 8) {
            int tm = (tt >> 2) * 16, tn = (tt & 3) * 16;
            FragC s;
#pragma unroll
            for (int r = 0; r < 8; ++r) s.f[r] = 0.f;
#pragma unroll
            for (int kc = 0; kc < 2; ++kc) {
                FragAB a, b;
#pragma unroll
                for (int v = 0; v < 8; ++v) {
                    int kpa = ((v < 4) ? 0 : 8) + hh * 4 + (v & 3);
                    a.u[v] = Qs[tm + am][kc * 16 + kpa];
                }
#pragma unroll
                for (int v = 0; v < 8; ++v) {
                    int kpb = ((v < 4) ? 0 : 8) + hh * 4 + (v & 3);
                    b.u[v] = Ks[tn + am][kc * 16 + kpb];
                }
                s.v = __builtin_amdgcn_wmma_f32_16x16x32_bf16(
                    false, a.v, false, b.v, (short)0, s.v, false, false);
            }
#pragma unroll
            for (int r = 0; r < 8; ++r) Sf[tm + r + 8 * hh][tn + am] = s.f[r];
        }
        __syncthreads();

        // online softmax, one thread per query row
        if (tid < QT) {
            int r = tid;
            float mold = mrow[r];
            float mx = mold;
            for (int c = 0; c < KT; ++c) mx = fmaxf(mx, Sf[r][c] * 0.125f);
            float scale = __expf(mold - mx);
            float sum = 0.f;
            for (int c = 0; c < KT; c += 2) {
                float p0 = __expf(Sf[r][c] * 0.125f - mx);
                float p1 = __expf(Sf[r][c + 1] * 0.125f - mx);
                sum += p0 + p1;
                Ps[r][c >> 1] = packbf2(p0, p1);
            }
            mrow[r] = mx;
            lrow[r] = lrow[r] * scale + sum;
            srow[r] = scale;
        }
        __syncthreads();

        // rescale running O, then O += P @ V
#pragma unroll
        for (int j = 0; j < 2; ++j)
#pragma unroll
            for (int r = 0; r < 8; ++r) o[j].f[r] *= srow[wr + r + 8 * hh];
#pragma unroll
        for (int kc = 0; kc < 2; ++kc) {
            FragAB a, b0, b1;
#pragma unroll
            for (int v = 0; v < 8; ++v) {
                int kpa = ((v < 4) ? 0 : 8) + hh * 4 + (v & 3);
                a.u[v] = Ps[wr + am][kc * 16 + kpa];
            }
#pragma unroll
            for (int v = 0; v < 8; ++v) {
                int kpb = ((v < 4) ? 0 : 8) + hh * 4 + (v & 3);
                b0.u[v] = Vt[wc + am][kc * 16 + kpb];
                b1.u[v] = Vt[wc + 16 + am][kc * 16 + kpb];
            }
            o[0].v = __builtin_amdgcn_wmma_f32_16x16x32_bf16(
                false, a.v, false, b0.v, (short)0, o[0].v, false, false);
            o[1].v = __builtin_amdgcn_wmma_f32_16x16x32_bf16(
                false, a.v, false, b1.v, (short)0, o[1].v, false, false);
        }
        __syncthreads();
    }

    // normalize and store bf16, token-major [token][h*64+d]
#pragma unroll
    for (int j = 0; j < 2; ++j)
#pragma unroll
        for (int r = 0; r < 8; ++r) {
            int row = wr + r + 8 * hh;
            int col = wc + j * 16 + am;
            float val = o[j].f[r] / lrow[row];
            Og[(size_t)(q0 + row) * HID + h * HD + col] = f2bf(val);
        }
}

// ---------------------------------------------------------------------------
// launcher
// ---------------------------------------------------------------------------
extern "C" void kernel_launch(void* const* d_in, const int* in_sizes, int n_in,
                              void* d_out, int out_size, void* d_ws, size_t ws_size,
                              hipStream_t stream) {
    (void)in_sizes; (void)n_in; (void)out_size; (void)ws_size;
    const float* img        = (const float*)d_in[0];
    const float* txt        = (const float*)d_in[1];
    const float* vec        = (const float*)d_in[2];
    const float* img_mod_w  = (const float*)d_in[3];
    const float* img_mod_b  = (const float*)d_in[4];
    const float* txt_mod_w  = (const float*)d_in[5];
    const float* txt_mod_b  = (const float*)d_in[6];
    const float* img_qkv_w  = (const float*)d_in[7];
    const float* img_q_scale= (const float*)d_in[8];
    const float* img_k_scale= (const float*)d_in[9];
    const float* img_proj_w = (const float*)d_in[10];
    const float* img_proj_b = (const float*)d_in[11];
    const float* img_mlp_w1 = (const float*)d_in[12];
    const float* img_mlp_b1 = (const float*)d_in[13];
    const float* img_mlp_w2 = (const float*)d_in[14];
    const float* img_mlp_b2 = (const float*)d_in[15];
    const float* txt_qkv_w  = (const float*)d_in[16];
    const float* txt_q_scale= (const float*)d_in[17];
    const float* txt_k_scale= (const float*)d_in[18];
    const float* txt_proj_w = (const float*)d_in[19];
    const float* txt_proj_b = (const float*)d_in[20];
    const float* txt_mlp_w1 = (const float*)d_in[21];
    const float* txt_mlp_b1 = (const float*)d_in[22];
    const float* txt_mlp_w2 = (const float*)d_in[23];
    const float* txt_mlp_b2 = (const float*)d_in[24];

    char* ws = (char*)d_ws;
    size_t off = 0;
    auto carve = [&](size_t bytes) -> void* {
        void* p = ws + off;
        off = (off + bytes + 255) & ~(size_t)255;
        return p;
    };
    float* i_mod = (float*)carve(6 * HID * sizeof(float));
    float* t_mod = (float*)carve(6 * HID * sizeof(float));
    unsigned short* xmod_img = (unsigned short*)carve((size_t)LIMG * HID * 2);
    unsigned short* xmod_txt = (unsigned short*)carve((size_t)LTXT * HID * 2);
    float* qkv_img = (float*)carve((size_t)LIMG * 3 * HID * 4);
    float* qkv_txt = (float*)carve((size_t)LTXT * 3 * HID * 4);
    unsigned short* Qg = (unsigned short*)carve((size_t)NH * LTOT * HD * 2);
    unsigned short* Kg = (unsigned short*)carve((size_t)NH * LTOT * HD * 2);
    unsigned short* Vg = (unsigned short*)carve((size_t)NH * LTOT * HD * 2);
    unsigned short* attn = (unsigned short*)carve((size_t)LTOT * HID * 2);
    float* img_mid = (float*)carve((size_t)LIMG * HID * 4);
    float* txt_mid = (float*)carve((size_t)LTXT * HID * 4);
    unsigned short* h_img = (unsigned short*)carve((size_t)LIMG * MLPD * 2);
    unsigned short* h_txt = (unsigned short*)carve((size_t)LTXT * MLPD * 2);
    // transposed bf16 weights [N][K]
    unsigned short* wt_img_qkv = (unsigned short*)carve((size_t)3 * HID * HID * 2);
    unsigned short* wt_txt_qkv = (unsigned short*)carve((size_t)3 * HID * HID * 2);
    unsigned short* wt_img_proj= (unsigned short*)carve((size_t)HID * HID * 2);
    unsigned short* wt_txt_proj= (unsigned short*)carve((size_t)HID * HID * 2);
    unsigned short* wt_img_w1  = (unsigned short*)carve((size_t)MLPD * HID * 2);
    unsigned short* wt_txt_w1  = (unsigned short*)carve((size_t)MLPD * HID * 2);
    unsigned short* wt_img_w2  = (unsigned short*)carve((size_t)HID * MLPD * 2);
    unsigned short* wt_txt_w2  = (unsigned short*)carve((size_t)HID * MLPD * 2);

    float* out_img = (float*)d_out;
    float* out_txt = (float*)d_out + (size_t)LIMG * HID;

    // 0) weight conversion (f32 [K][N] -> bf16 [N][K])
    convert_transpose<<<dim3(HID/32, 3*HID/32), 256, 0, stream>>>(img_qkv_w, wt_img_qkv, HID, 3*HID);
    convert_transpose<<<dim3(HID/32, 3*HID/32), 256, 0, stream>>>(txt_qkv_w, wt_txt_qkv, HID, 3*HID);
    convert_transpose<<<dim3(HID/32, HID/32),  256, 0, stream>>>(img_proj_w, wt_img_proj, HID, HID);
    convert_transpose<<<dim3(HID/32, HID/32),  256, 0, stream>>>(txt_proj_w, wt_txt_proj, HID, HID);
    convert_transpose<<<dim3(HID/32, MLPD/32), 256, 0, stream>>>(img_mlp_w1, wt_img_w1, HID, MLPD);
    convert_transpose<<<dim3(HID/32, MLPD/32), 256, 0, stream>>>(txt_mlp_w1, wt_txt_w1, HID, MLPD);
    convert_transpose<<<dim3(MLPD/32, HID/32), 256, 0, stream>>>(img_mlp_w2, wt_img_w2, MLPD, HID);
    convert_transpose<<<dim3(MLPD/32, HID/32), 256, 0, stream>>>(txt_mlp_w2, wt_txt_w2, MLPD, HID);

    // 1) modulation vectors
    mod_gemv<<<24, 256, 0, stream>>>(vec, img_mod_w, img_mod_b, i_mod, 6 * HID);
    mod_gemv<<<24, 256, 0, stream>>>(vec, txt_mod_w, txt_mod_b, t_mod, 6 * HID);

    // 2) LN + modulate (sh1=0, sc1=1)
    ln_mod<<<LIMG, 256, 0, stream>>>(img, i_mod, 0, 1, xmod_img);
    ln_mod<<<LTXT, 256, 0, stream>>>(txt, t_mod, 0, 1, xmod_txt);

    // 3) QKV GEMMs
    gemm_bf16<<<dim3(LIMG / BM, 3 * HID / BN), 256, 0, stream>>>(
        xmod_img, wt_img_qkv, nullptr, nullptr, nullptr, qkv_img, LIMG, 3 * HID, HID, 0);
    gemm_bf16<<<dim3(LTXT / BM, 3 * HID / BN), 256, 0, stream>>>(
        xmod_txt, wt_txt_qkv, nullptr, nullptr, nullptr, qkv_txt, LTXT, 3 * HID, HID, 0);

    // 4) RMS norm + head-major pack (txt tokens first, concat order)
    qkv_post<<<(LTXT * NH) / 8, 256, 0, stream>>>(qkv_txt, txt_q_scale, txt_k_scale,
                                                  Qg, Kg, Vg, LTXT, 0);
    qkv_post<<<(LIMG * NH) / 8, 256, 0, stream>>>(qkv_img, img_q_scale, img_k_scale,
                                                  Qg, Kg, Vg, LIMG, LTXT);

    // 5) joint attention
    attn_kernel<<<dim3(LTOT / QT, NH), 256, 0, stream>>>(Qg, Kg, Vg, attn);

    // 6) out projection + gate*x + residual (g1 = mod[2])
    gemm_bf16<<<dim3(LIMG / BM, HID / BN), 256, 0, stream>>>(
        attn + (size_t)LTXT * HID, wt_img_proj, img_proj_b, img, i_mod + 2 * HID,
        img_mid, LIMG, HID, HID, 2);
    gemm_bf16<<<dim3(LTXT / BM, HID / BN), 256, 0, stream>>>(
        attn, wt_txt_proj, txt_proj_b, txt, t_mod + 2 * HID,
        txt_mid, LTXT, HID, HID, 2);

    // 7) LN + modulate (sh2=3, sc2=4)
    ln_mod<<<LIMG, 256, 0, stream>>>(img_mid, i_mod, 3, 4, xmod_img);
    ln_mod<<<LTXT, 256, 0, stream>>>(txt_mid, t_mod, 3, 4, xmod_txt);

    // 8) MLP up-proj + GELU -> bf16
    gemm_bf16<<<dim3(LIMG / BM, MLPD / BN), 256, 0, stream>>>(
        xmod_img, wt_img_w1, img_mlp_b1, nullptr, nullptr, h_img, LIMG, MLPD, HID, 1);
    gemm_bf16<<<dim3(LTXT / BM, MLPD / BN), 256, 0, stream>>>(
        xmod_txt, wt_txt_w1, txt_mlp_b1, nullptr, nullptr, h_txt, LTXT, MLPD, HID, 1);

    // 9) MLP down-proj + gate*x + residual -> final outputs (g2 = mod[5])
    gemm_bf16<<<dim3(LIMG / BM, HID / BN), 256, 0, stream>>>(
        h_img, wt_img_w2, img_mlp_b2, img_mid, i_mod + 5 * HID, out_img, LIMG, HID, MLPD, 2);
    gemm_bf16<<<dim3(LTXT / BM, HID / BN), 256, 0, stream>>>(
        h_txt, wt_txt_w2, txt_mlp_b2, txt_mid, t_mod + 5 * HID, out_txt, LTXT, HID, MLPD, 2);
}
